// TransitionDown_1984274891515
// MI455X (gfx1250) — compile-verified, matrix operand order checked by reading, not run on hardware
//
#include <hip/hip_runtime.h>
#include <hip/hip_bf16.h>
#include <cstdint>
#include <cstddef>

// ---------------- problem constants (match reference) ----------------
#define BGRAPH 8
#define NPG    4096
#define CIN    256
#define COUT   512
#define KNN    16
#define KNB    (KNN + 1)     // neighbors incl. self
#define MSAMP  1024          // NPG * 0.25
#define EPSV   1e-5f

typedef __bf16 bf16_t;
typedef __attribute__((ext_vector_type(16))) __bf16 v16bf;
typedef __attribute__((ext_vector_type(8)))  float  v8f;

// ---------------- workspace layout (bytes) ----------------
static const size_t IDX_OFF  = 0;                                   // B*M int
static const size_t NBR_OFF  = IDX_OFF  + (size_t)BGRAPH*MSAMP*4;   // B*M*17 int
static const size_t MEAN_OFF = NBR_OFF  + (size_t)BGRAPH*MSAMP*KNB*4;
static const size_t RSTD_OFF = MEAN_OFF + (size_t)BGRAPH*COUT*4;
static const size_t XB_OFF   = RSTD_OFF + (size_t)BGRAPH*COUT*4;    // bf16 x
static const size_t WB_OFF   = XB_OFF   + (size_t)BGRAPH*NPG*CIN*2; // bf16 W frags
static const size_t H_OFF    = WB_OFF   + (size_t)CIN*COUT*2;       // fp32 h
static const size_t WS_TOTAL = H_OFF    + (size_t)BGRAPH*NPG*COUT*4;

// ---------------- 1) fp32 -> bf16 convert of x ----------------
__global__ __launch_bounds__(256)
void convert_x_kernel(const float* __restrict__ x, bf16_t* __restrict__ xb, int n)
{
    int i = (blockIdx.x * blockDim.x + threadIdx.x) * 4;
    if (i >= n) return;
    float4 f = *reinterpret_cast<const float4*>(x + i);
    xb[i + 0] = (bf16_t)f.x;
    xb[i + 1] = (bf16_t)f.y;
    xb[i + 2] = (bf16_t)f.z;
    xb[i + 3] = (bf16_t)f.w;
}

// ---------------- 2) pre-swizzle W into per-lane B fragments ----------------
// Fragment layout for V_WMMA_F32_16X16X32_BF16 B (32x16, K x N), mirroring the
// documented 16-bit A layout: lane<16 holds K {0..7,16..23}, lane>=16 holds
// K {8..15,24..31}; column N = lane%16. One fragment = 32 lanes * 16 bf16.
__global__ __launch_bounds__(256)
void prep_w_kernel(const float* __restrict__ W, bf16_t* __restrict__ wfrag)
{
    int tid = blockIdx.x * blockDim.x + threadIdx.x;
    if (tid >= (CIN / 32) * 32 * 32 * 16) return;   // 8 ktiles * 32 ntiles * 32 lanes * 16
    int e    = tid & 15;
    int lane = (tid >> 4) & 31;
    int nt   = (tid >> 9) & 31;
    int kt   = tid >> 14;
    int half = lane >> 4;
    int kk   = (e < 8) ? (half * 8 + e) : (16 + half * 8 + (e - 8));
    int k    = kt * 32 + kk;
    int n    = nt * 16 + (lane & 15);
    wfrag[tid] = (bf16_t)W[(size_t)k * COUT + n];
}

// ---------------- 3) farthest point sampling, one block per graph ----------------
__global__ __launch_bounds__(256)
void fps_kernel(const float* __restrict__ pos, int* __restrict__ idx,
                float* __restrict__ subpos, float* __restrict__ subbatch)
{
    __shared__ float dist[NPG];     // 16 KB
    __shared__ float sval[256];
    __shared__ int   sidx[256];
    __shared__ float cpos[3];

    const int b = blockIdx.x;
    const int t = threadIdx.x;
    const float* pg = pos + (size_t)b * NPG * 3;

    const float p0x = pg[0], p0y = pg[1], p0z = pg[2];
    for (int i = t; i < NPG; i += 256) {
        float dx = pg[i * 3 + 0] - p0x;
        float dy = pg[i * 3 + 1] - p0y;
        float dz = pg[i * 3 + 2] - p0z;
        dist[i] = dx * dx + dy * dy + dz * dz;
    }
    if (t == 0) idx[b * MSAMP + 0] = 0;
    __syncthreads();

    for (int m = 1; m < MSAMP; ++m) {
        // block argmax (tie-break: lowest index, matching jnp.argmax)
        float bv = -1.0f; int bi = 0x7fffffff;
        for (int i = t; i < NPG; i += 256) {
            float v = dist[i];
            if (v > bv) { bv = v; bi = i; }
        }
        sval[t] = bv; sidx[t] = bi;
        __syncthreads();
        for (int s = 128; s > 0; s >>= 1) {
            if (t < s) {
                float v2 = sval[t + s]; int i2 = sidx[t + s];
                if (v2 > sval[t] || (v2 == sval[t] && i2 < sidx[t])) { sval[t] = v2; sidx[t] = i2; }
            }
            __syncthreads();
        }
        if (t == 0) {
            int ci = sidx[0];
            idx[b * MSAMP + m] = ci;
            cpos[0] = pg[ci * 3 + 0];
            cpos[1] = pg[ci * 3 + 1];
            cpos[2] = pg[ci * 3 + 2];
        }
        __syncthreads();
        const float cx = cpos[0], cy = cpos[1], cz = cpos[2];
        for (int i = t; i < NPG; i += 256) {
            float dx = pg[i * 3 + 0] - cx;
            float dy = pg[i * 3 + 1] - cy;
            float dz = pg[i * 3 + 2] - cz;
            float nd = dx * dx + dy * dy + dz * dz;
            if (nd < dist[i]) dist[i] = nd;
        }
        __syncthreads();
    }

    // emit sub_pos / sub_batch (batch value stored as float)
    for (int m = t; m < MSAMP; m += 256) {
        int ci = idx[b * MSAMP + m];
        int s  = b * MSAMP + m;
        subpos[s * 3 + 0] = pg[ci * 3 + 0];
        subpos[s * 3 + 1] = pg[ci * 3 + 1];
        subpos[s * 3 + 2] = pg[ci * 3 + 2];
        subbatch[s] = (float)b;
    }
}

// ---------------- 4) KNN: thread per sampled point, 17-deep insertion ----------------
__global__ __launch_bounds__(256)
void knn_kernel(const float* __restrict__ pos, const int* __restrict__ idx,
                int* __restrict__ nbr)
{
    int s = blockIdx.x * blockDim.x + threadIdx.x;
    if (s >= BGRAPH * MSAMP) return;
    int b = s / MSAMP;
    const float* pg = pos + (size_t)b * NPG * 3;
    int ci = idx[s];
    const float qx = pg[ci * 3 + 0], qy = pg[ci * 3 + 1], qz = pg[ci * 3 + 2];

    float bd[KNB]; int bi[KNB];
    for (int j = 0; j < KNB; ++j) { bd[j] = 1e30f; bi[j] = 0; }

    for (int n = 0; n < NPG; ++n) {
        float dx = pg[n * 3 + 0] - qx;
        float dy = pg[n * 3 + 1] - qy;
        float dz = pg[n * 3 + 2] - qz;
        float d2 = dx * dx + dy * dy + dz * dz;
        if (d2 < bd[KNB - 1]) {
            int j = KNB - 1;
            while (j > 0 && bd[j - 1] > d2) { bd[j] = bd[j - 1]; bi[j] = bi[j - 1]; --j; }
            bd[j] = d2; bi[j] = n;
        }
    }
    for (int j = 0; j < KNB; ++j) nbr[(size_t)s * KNB + j] = bi[j];
}

// ---------------- 5) h = x @ W + b via bf16 WMMA, fp32 accumulate ----------------
// One wave per (32-row M stripe) x (64-col N group): 2 M-tiles x 4 N-tiles.
// Each loaded B fragment feeds two independent WMMAs (two M-tiles), halving
// B traffic and giving the scheduler parallel MMA chains to overlap loads.
__global__ __launch_bounds__(256)
void gemm_wmma_kernel(const bf16_t* __restrict__ xb,
                      const bf16_t* __restrict__ wfrag,
                      const float* __restrict__ bias,
                      float* __restrict__ h)
{
    const int lane  = threadIdx.x & 31;
    const int wave  = threadIdx.x >> 5;   // N group 0..7 (64 cols each)
    const int tileM = blockIdx.x;         // 0..1023, 32 rows each
    const int half  = lane >> 4;
    const int l16   = lane & 15;

    const bf16_t* arow0 = xb + ((size_t)tileM * 32 + l16) * CIN;
    const bf16_t* arow1 = arow0 + (size_t)16 * CIN;

    v8f acc[2][4] = {};

#pragma unroll
    for (int kt = 0; kt < CIN / 32; ++kt) {
        // A fragments: lane<16 -> K {base..base+7, base+16..base+23}; lane>=16 +8
        v16bf a0, a1;
        uint4* ap0 = reinterpret_cast<uint4*>(&a0);
        uint4* ap1 = reinterpret_cast<uint4*>(&a1);
        const bf16_t* aptr0 = arow0 + kt * 32 + half * 8;
        const bf16_t* aptr1 = arow1 + kt * 32 + half * 8;
        ap0[0] = *reinterpret_cast<const uint4*>(aptr0);
        ap0[1] = *reinterpret_cast<const uint4*>(aptr0 + 16);
        ap1[0] = *reinterpret_cast<const uint4*>(aptr1);
        ap1[1] = *reinterpret_cast<const uint4*>(aptr1 + 16);
        if (kt + 1 < CIN / 32) {
            __builtin_prefetch(arow0 + (kt + 1) * 32, 0, 1);   // global_prefetch_b8
            __builtin_prefetch(arow1 + (kt + 1) * 32, 0, 1);
        }

#pragma unroll
        for (int t = 0; t < 4; ++t) {
            int nt = wave * 4 + t;
            const bf16_t* bptr = wfrag + (((size_t)kt * 32 + nt) * 32 + lane) * 16;
            v16bf bm = *reinterpret_cast<const v16bf*>(bptr);
            acc[0][t] = __builtin_amdgcn_wmma_f32_16x16x32_bf16(
                false, a0, false, bm, (short)0, acc[0][t], false, false);
            acc[1][t] = __builtin_amdgcn_wmma_f32_16x16x32_bf16(
                false, a1, false, bm, (short)0, acc[1][t], false, false);
        }
    }

    // C/D layout: lane<16 -> N=lane, rows r + 0; lane>=16 -> rows r + 8
#pragma unroll
    for (int mi = 0; mi < 2; ++mi) {
        int rbase = tileM * 32 + mi * 16 + half * 8;
#pragma unroll
        for (int t = 0; t < 4; ++t) {
            int n = wave * 64 + t * 16 + l16;
            float bb = bias[n];
#pragma unroll
            for (int r = 0; r < 8; ++r)
                h[(size_t)(rbase + r) * COUT + n] = acc[mi][t][r] + bb;
        }
    }
}

// ---------------- 6) per (graph, channel) mean / rstd ----------------
__global__ __launch_bounds__(512)
void stats_kernel(const float* __restrict__ h, float* __restrict__ mean,
                  float* __restrict__ rstd)
{
    int b = blockIdx.x, c = threadIdx.x;
    const float* hb = h + (size_t)b * NPG * COUT + c;
    float s = 0.f, sq = 0.f;
    for (int r = 0; r < NPG; ++r) {
        float v = hb[(size_t)r * COUT];
        s += v; sq += v * v;
    }
    float m   = s * (1.0f / NPG);
    float var = sq * (1.0f / NPG) - m * m;
    mean[b * COUT + c] = m;
    rstd[b * COUT + c] = rsqrtf(var + EPSV);
}

// ---------------- 7) gather-max over 17 neighbors, then norm + relu ----------------
// relu((x-mean)*rstd) with rstd>0 commutes with max, so max raw h first.
__global__ __launch_bounds__(512)
void gather_kernel(const float* __restrict__ h, const int* __restrict__ nbr,
                   const float* __restrict__ mean, const float* __restrict__ rstd,
                   float* __restrict__ out)
{
    __shared__ int sn[KNB];
    int s = blockIdx.x;            // sampled point 0..8191
    int b = s / MSAMP;
    int c = threadIdx.x;           // channel
    if (c < KNB) sn[c] = nbr[(size_t)s * KNB + c];
    __syncthreads();

    const float* hb = h + (size_t)b * NPG * COUT + c;
    float mx = -1e30f;
#pragma unroll
    for (int j = 0; j < KNB; ++j)
        mx = fmaxf(mx, hb[(size_t)sn[j] * COUT]);

    float v = (mx - mean[b * COUT + c]) * rstd[b * COUT + c];
    out[(size_t)s * COUT + c] = fmaxf(v, 0.f);
}

// ---------------- launch ----------------
extern "C" void kernel_launch(void* const* d_in, const int* in_sizes, int n_in,
                              void* d_out, int out_size, void* d_ws, size_t ws_size,
                              hipStream_t stream)
{
    const float* x   = (const float*)d_in[0];
    const float* pos = (const float*)d_in[1];
    // d_in[2] = batch (int32): synthesized analytically (graph b -> value b)
    const float* W   = (const float*)d_in[3];
    const float* bv  = (const float*)d_in[4];

    if (ws_size < WS_TOTAL) return;

    char* ws = (char*)d_ws;
    int*    idx   = (int*)   (ws + IDX_OFF);
    int*    nbr   = (int*)   (ws + NBR_OFF);
    float*  mean  = (float*) (ws + MEAN_OFF);
    float*  rstd  = (float*) (ws + RSTD_OFF);
    bf16_t* xb    = (bf16_t*)(ws + XB_OFF);
    bf16_t* wfrag = (bf16_t*)(ws + WB_OFF);
    float*  h     = (float*) (ws + H_OFF);

    float* out      = (float*)d_out;
    float* subpos   = out + (size_t)BGRAPH * MSAMP * COUT;
    float* subbatch = subpos + (size_t)BGRAPH * MSAMP * 3;

    const int nx = BGRAPH * NPG * CIN;                       // 8,388,608
    convert_x_kernel<<<nx / (256 * 4), 256, 0, stream>>>(x, xb, nx);
    prep_w_kernel   <<<(CIN / 32) * 32 * 32 * 16 / 256, 256, 0, stream>>>(W, wfrag);
    fps_kernel      <<<BGRAPH, 256, 0, stream>>>(pos, idx, subpos, subbatch);
    knn_kernel      <<<(BGRAPH * MSAMP + 255) / 256, 256, 0, stream>>>(pos, idx, nbr);
    gemm_wmma_kernel<<<(BGRAPH * NPG) / 32, 256, 0, stream>>>(xb, wfrag, bv, h);
    stats_kernel    <<<BGRAPH, 512, 0, stream>>>(h, mean, rstd);
    gather_kernel   <<<BGRAPH * MSAMP, 512, 0, stream>>>(h, nbr, mean, rstd, out);
}
